// DecoderStory_71880572666639
// MI455X (gfx1250) — matrix operand order repaired and checked by reading, not compiled, not run on hardware
//
#include <hip/hip_runtime.h>
#include <hip/hip_bf16.h>

// Problem constants
#define B_    64
#define T_    32
#define E_    256
#define H_    512
#define V_    10000
#define NSEQ  2048      // B*T, length of the flattened scan
#define D1_   768       // E + H
#define G4    2048      // 4*H gate rows
#define VPAD  10048     // V padded to multiple of 64 for WMMA tiling
#define NWG   32        // persistent workgroups for the scan
#define STPB  256       // threads per scan workgroup

typedef __attribute__((ext_vector_type(16))) __bf16 v16bf;
typedef __attribute__((ext_vector_type(8)))  float  v8f;
typedef __attribute__((ext_vector_type(4)))  unsigned v4u;
typedef __attribute__((ext_vector_type(8)))  int    v8i;
typedef __attribute__((ext_vector_type(4)))  int    v4i;
typedef unsigned short u16;
typedef unsigned int   u32;

#if defined(__gfx1250__) && __has_builtin(__builtin_amdgcn_tensor_load_to_lds) && \
    __has_builtin(__builtin_amdgcn_s_wait_tensorcnt)
#define USE_TDM 1
#else
#define USE_TDM 0
#endif

union BfTile { uint4 u[2]; v16bf v; };

__device__ inline u16 f2bf(float f) {               // round-to-nearest-even f32->bf16
  u32 u = __float_as_uint(f);
  return (u16)((u + 0x7FFFu + ((u >> 16) & 1u)) >> 16);
}
__device__ inline float bflo(u32 p) { return __uint_as_float(p << 16); }
__device__ inline float bfhi(u32 p) { return __uint_as_float(p & 0xFFFF0000u); }
__device__ inline float sigm(float x) { return 1.f / (1.f + __expf(-x)); }

// Monotonic-target grid barrier (device-scope atomic counter).
__device__ inline void grid_sync(u32* cnt, u32 target) {
  __syncthreads();
  if (threadIdx.x == 0) {
    __hip_atomic_fetch_add(cnt, 1u, __ATOMIC_ACQ_REL, __HIP_MEMORY_SCOPE_AGENT);
    while (__hip_atomic_load(cnt, __ATOMIC_ACQUIRE, __HIP_MEMORY_SCOPE_AGENT) < target)
      __builtin_amdgcn_s_sleep(1);
  }
  __syncthreads();
}

// ---------------- small prep kernels ----------------

__global__ void start_rows_kernel(float* out) {     // out[b,0,:] = start vec
  int idx = blockIdx.x * 256 + threadIdx.x;
  if (idx >= B_ * V_) return;
  int b = idx / V_, v = idx - b * V_;
  out[(size_t)b * T_ * V_ + v] = (v == 1) ? 10000.0f : 0.0f;
}

__global__ void init_sync_kernel(u32* cnt, float* h1g, float* h2g) {
  if (threadIdx.x == 0) *cnt = 0u;
  for (int i = threadIdx.x; i < 2 * H_; i += 256) { h1g[i] = 0.f; h2g[i] = 0.f; }
}

__global__ void feats_kernel(const float* __restrict__ story, const float* __restrict__ Wst,
                             const float* __restrict__ bst, float* __restrict__ feats) {
  int idx = blockIdx.x * 256 + threadIdx.x;         // exactly B_*H_ threads
  int b = idx >> 9, h = idx & 511;
  const float* sp = story + (size_t)b * (2 * H_);
  const float* wp = Wst   + (size_t)h * (2 * H_);
  float s = bst[h];
  for (int k = 0; k < 2 * H_; ++k) s = fmaf(sp[k], wp[k], s);
  feats[idx] = fmaxf(s, 0.f);
}

// x[n] = concat(feats[b], embed[captions[n]]) in bf16
__global__ void build_x_kernel(const float* __restrict__ feats, const float* __restrict__ embed,
                               const int* __restrict__ captions, u16* __restrict__ Xbf) {
  int idx = blockIdx.x * 256 + threadIdx.x;         // NSEQ*D1_
  int n = idx / D1_, k = idx - n * D1_;
  float v = (k < H_) ? feats[(n >> 5) * H_ + k]
                     : embed[(size_t)captions[n] * E_ + (k - H_)];
  Xbf[idx] = f2bf(v);
}

__global__ void cvt_bf_kernel(const float* __restrict__ src, u16* __restrict__ dst, int count) {
  int idx = blockIdx.x * 256 + threadIdx.x;
  if (idx < count) dst[idx] = f2bf(src[idx]);
}

__global__ void wout_pad_kernel(const float* __restrict__ Wout, u16* __restrict__ dst) {
  int idx = blockIdx.x * 256 + threadIdx.x;         // VPAD*H_
  int v = idx >> 9, k = idx & 511;
  dst[idx] = (v < V_) ? f2bf(Wout[(size_t)v * H_ + k]) : (u16)0;
}

// ---------------- WMMA GEMM: P1 = X * W_ih1^T + b1  (M=2048,N=2048,K=768) ----------------
// One wave computes a 32(M) x 64(N) tile: 8 accumulators, A0/A1 reused across 4 B tiles
// (12 b128 loads : 8 WMMAs per K-step).
__global__ void proj1_wmma_kernel(const u16* __restrict__ A, const u16* __restrict__ Bm,
                                  const float* __restrict__ bias, float* __restrict__ D) {
  int l = threadIdx.x;
  int m0 = blockIdx.y * 32, n0 = blockIdx.x * 64;
  int rlo = l & 15, hi = l >> 4;
  v8f acc[2][4] = {};
  const u16* arow0 = A  + (size_t)(m0 + rlo) * D1_ + hi * 8;   // A: K split {0-7,16-23}/{8-15,24-31}
  const u16* arow1 = arow0 + (size_t)16 * D1_;
  const u16* brow0 = Bm + (size_t)(n0 + rlo) * D1_ + hi * 8;   // B: K split {0-7,16-23}/{8-15,24-31}
  for (int k0 = 0; k0 < D1_; k0 += 32) {
    BfTile a0, a1;
    a0.u[0] = *(const uint4*)(arow0 + k0);
    a0.u[1] = *(const uint4*)(arow0 + k0 + 16);
    a1.u[0] = *(const uint4*)(arow1 + k0);
    a1.u[1] = *(const uint4*)(arow1 + k0 + 16);
#pragma unroll
    for (int j = 0; j < 4; ++j) {
      BfTile bt;
      const u16* bp = brow0 + (size_t)j * 16 * D1_ + k0;
      bt.u[0] = *(const uint4*)(bp);
      bt.u[1] = *(const uint4*)(bp + 16);
      acc[0][j] = __builtin_amdgcn_wmma_f32_16x16x32_bf16(false, a0.v, false, bt.v,
                                                          (short)0, acc[0][j], false, false);
      acc[1][j] = __builtin_amdgcn_wmma_f32_16x16x32_bf16(false, a1.v, false, bt.v,
                                                          (short)0, acc[1][j], false, false);
    }
  }
#pragma unroll
  for (int mt = 0; mt < 2; ++mt) {
#pragma unroll
    for (int j = 0; j < 4; ++j) {
      int col = n0 + j * 16 + rlo;
      float bv = bias[col];
#pragma unroll
      for (int g = 0; g < 8; ++g) {
        int row = m0 + mt * 16 + (hi ? 8 : 0) + g;
        D[(size_t)row * G4 + col] = acc[mt][j][g] + bv;
      }
    }
  }
}

// ---------------- WMMA GEMM: logits = Y * W_out^T + b_out, shifted/masked store ----------------
__global__ void logits_wmma_kernel(const u16* __restrict__ A, const u16* __restrict__ Bm,
                                   const float* __restrict__ bias, const int* __restrict__ lengths,
                                   float* __restrict__ out) {
  int l = threadIdx.x;
  int m0 = blockIdx.y * 32, n0 = blockIdx.x * 64;
  int rlo = l & 15, hi = l >> 4;
  v8f acc[2][4] = {};
  const u16* arow0 = A  + (size_t)(m0 + rlo) * H_ + hi * 8;
  const u16* arow1 = arow0 + (size_t)16 * H_;
  const u16* brow0 = Bm + (size_t)(n0 + rlo) * H_ + hi * 8;
  for (int k0 = 0; k0 < H_; k0 += 32) {
    BfTile a0, a1;
    a0.u[0] = *(const uint4*)(arow0 + k0);
    a0.u[1] = *(const uint4*)(arow0 + k0 + 16);
    a1.u[0] = *(const uint4*)(arow1 + k0);
    a1.u[1] = *(const uint4*)(arow1 + k0 + 16);
#pragma unroll
    for (int j = 0; j < 4; ++j) {
      BfTile bt;
      const u16* bp = brow0 + (size_t)j * 16 * H_ + k0;
      bt.u[0] = *(const uint4*)(bp);
      bt.u[1] = *(const uint4*)(bp + 16);
      acc[0][j] = __builtin_amdgcn_wmma_f32_16x16x32_bf16(false, a0.v, false, bt.v,
                                                          (short)0, acc[0][j], false, false);
      acc[1][j] = __builtin_amdgcn_wmma_f32_16x16x32_bf16(false, a1.v, false, bt.v,
                                                          (short)0, acc[1][j], false, false);
    }
  }
#pragma unroll
  for (int mt = 0; mt < 2; ++mt) {
#pragma unroll
    for (int j = 0; j < 4; ++j) {
      int col = n0 + j * 16 + rlo;
      if (col >= V_) continue;
      float bv = bias[col];
#pragma unroll
      for (int g = 0; g < 8; ++g) {
        int row = m0 + mt * 16 + (hi ? 8 : 0) + g;  // flattened (b,t)
        int t = row & 31;
        if (t == T_ - 1) continue;                   // last step's logits are dropped
        int b = row >> 5;
        bool valid = t < (lengths[b] - 1);
        out[((size_t)b * T_ + (t + 1)) * V_ + col] = valid ? (acc[mt][j][g] + bv) : 0.f;
      }
    }
  }
}

// ---------------- persistent 2-layer LSTM scan over 2048 sequential steps ----------------
// 32 WGs; WG w owns hidden slice j in [16w,16w+16) and gate rows {j, j+512, j+1024, j+1536}
// of W_hh1 / W_ih2 / W_hh2, resident in LDS (3*64KB bf16). 2 grid barriers per step.
// Weight preload uses the Tensor Data Mover: one 3D-tile descriptor per weight matrix
// (x: 512 contiguous bf16, y: 16 rows stride 512, z: 4 gate blocks stride 512*512),
// landing contiguously in LDS. Dynamic LDS base offset is 0 (no static LDS in this TU).
__global__ void __launch_bounds__(STPB, 1)
lstm_scan_kernel(const float* __restrict__ P1,
                 const u16* __restrict__ Whh1, const u16* __restrict__ Wih2,
                 const u16* __restrict__ Whh2, const float* __restrict__ b2,
                 const int* __restrict__ lengths,
                 float* __restrict__ h1g, float* __restrict__ h2g,
                 u16* __restrict__ Ybf, u32* __restrict__ cnt) {
  extern __shared__ unsigned char smem[];
  u16*  w1  = (u16*)smem;                 // 64 rows x 512 bf16 (LDS offset 0)
  u16*  w2i = w1  + 64 * 512;             // LDS offset 65536
  u16*  w2h = w2i + 64 * 512;             // LDS offset 131072
  float* h1s = (float*)(w2h + 64 * 512);  // 512 f32
  float* h2s = h1s + 512;                 // 512 f32
  float* gv  = h2s + 512;                 // 64  f32
  float* b2s = gv  + 64;                  // 64  f32

  const int tid = threadIdx.x;
  const int wg  = blockIdx.x;
  const int ridx = tid >> 2;              // local gate row 0..63
  const int qq   = tid & 3;               // quarter of the dot product
  const int rglob = (ridx >> 4) * H_ + wg * 16 + (ridx & 15);

#if USE_TDM
  if (tid < 32) {                         // one wave issues the DMAs (TDM ignores EXEC)
    auto tdm_load = [](const u16* gptr, unsigned lds_off) {
      unsigned long long ga = (unsigned long long)(size_t)gptr;
      v4u g0;                             // D# group 0
      g0[0] = 1u;                         // count=1, user descriptor
      g0[1] = lds_off;                    // lds_addr (bytes)
      g0[2] = (unsigned)(ga & 0xFFFFFFFFu);             // global_addr[31:0]
      g0[3] = (unsigned)((ga >> 32) & 0x01FFFFFFu)      // global_addr[56:32]
            | 0x80000000u;                              // type=2 ("image")
      v8i g1;                             // D# group 1
      g1[0] = 0x00010000;                 // wg_mask=0, data_size=1 (2 bytes)
      g1[1] = (int)(512u << 16);          // tensor_dim0[15:0]=512
      g1[2] = (int)(2048u << 16);         // tensor_dim0[31:16]=0, tensor_dim1[15:0]=2048
      g1[3] = (int)(512u << 16);          // tensor_dim1[31:16]=0, tile_dim0=512
      g1[4] = (int)(16u | (4u << 16));    // tile_dim1=16, tile_dim2=4
      g1[5] = 512;                        // tensor_dim0_stride[31:0]=512
      g1[6] = 0;                          // stride0[47:32]=0, stride1[15:0]=0 (262144&0xFFFF)
      g1[7] = 4;                          // tensor_dim1_stride[47:16]=4 -> 262144 elements
      v4i g2; g2[0] = 4; g2[1] = 0; g2[2] = 0; g2[3] = 0;  // tensor_dim2=4, no iterate
      v4i g3; g3[0] = 0; g3[1] = 0; g3[2] = 0; g3[3] = 0;
      v8i gx; gx[0]=0; gx[1]=0; gx[2]=0; gx[3]=0; gx[4]=0; gx[5]=0; gx[6]=0; gx[7]=0;
      // 6-arg toolchain variant: (group0, group1, group2, group3, extra, cpol)
      __builtin_amdgcn_tensor_load_to_lds(g0, g1, g2, g3, gx, 0);
    };
    tdm_load(Whh1 + (size_t)(wg * 16) * H_, 0u);
    tdm_load(Wih2 + (size_t)(wg * 16) * H_, 64u * 1024u);
    tdm_load(Whh2 + (size_t)(wg * 16) * H_, 128u * 1024u);
    __builtin_amdgcn_s_wait_tensorcnt(0);
  }
#else
  // Fallback: 256 threads copy 256 dwords per 512-elem row.
  for (int r = 0; r < 64; ++r) {
    int rg = (r >> 4) * H_ + wg * 16 + (r & 15);
    ((u32*)(w1  + r * 512))[tid] = ((const u32*)(Whh1 + (size_t)rg * H_))[tid];
    ((u32*)(w2i + r * 512))[tid] = ((const u32*)(Wih2 + (size_t)rg * H_))[tid];
    ((u32*)(w2h + r * 512))[tid] = ((const u32*)(Whh2 + (size_t)rg * H_))[tid];
  }
#endif
  if (tid < 64) b2s[tid] = b2[(tid >> 4) * H_ + wg * 16 + (tid & 15)];
  __syncthreads();

  float c1 = 0.f, c2 = 0.f;               // cell state slice lives in registers (tid<16)
  u32 bar = 0;

  for (int n = 0; n < NSEQ; ++n) {
    const int pb = (n & 1) * H_;          // state entering step n
    const int qb = ((n & 1) ^ 1) * H_;    // state leaving step n (double buffer)

    // ---- layer 1: g1 = P1[n] + Whh1 @ h1 ; gates ; h1 update ----
    for (int i = tid; i < H_; i += STPB) h1s[i] = h1g[pb + i];
    __syncthreads();
    {
      const u16* wr = w1 + ridx * 512 + qq * 128;
      const float* hh = h1s + qq * 128;
      float s = 0.f;
      for (int k = 0; k < 128; k += 8) {
        uint4 pk = *(const uint4*)(wr + k);
        s = fmaf(bflo(pk.x), hh[k+0], s); s = fmaf(bfhi(pk.x), hh[k+1], s);
        s = fmaf(bflo(pk.y), hh[k+2], s); s = fmaf(bfhi(pk.y), hh[k+3], s);
        s = fmaf(bflo(pk.z), hh[k+4], s); s = fmaf(bfhi(pk.z), hh[k+5], s);
        s = fmaf(bflo(pk.w), hh[k+6], s); s = fmaf(bfhi(pk.w), hh[k+7], s);
      }
      s += __shfl_xor(s, 1); s += __shfl_xor(s, 2);
      if (qq == 0) gv[ridx] = s + P1[(size_t)n * G4 + rglob];
    }
    __syncthreads();
    if (tid < 16) {
      int j = wg * 16 + tid;
      float gi = gv[tid], gf = gv[16 + tid], gc = gv[32 + tid], go = gv[48 + tid];
      float c1n = sigm(gf) * c1 + sigm(gi) * tanhf(gc);
      float h1n = sigm(go) * tanhf(c1n);
      bool m = (n & 31) < (lengths[n >> 5] - 1);
      c1 = m ? c1n : c1;
      h1g[qb + j] = m ? h1n : h1s[j];
    }
    __threadfence();
    bar += NWG; grid_sync(cnt, bar);

    // ---- layer 2: g2 = Wih2 @ h1_new + Whh2 @ h2 + b2 ; gates ; h2 update ----
    for (int i = tid; i < H_; i += STPB) { h1s[i] = h1g[qb + i]; h2s[i] = h2g[pb + i]; }
    __syncthreads();
    {
      const u16* wi = w2i + ridx * 512 + qq * 128;
      const u16* wh = w2h + ridx * 512 + qq * 128;
      const float* ha = h1s + qq * 128;
      const float* hb = h2s + qq * 128;
      float s = 0.f;
      for (int k = 0; k < 128; k += 8) {
        uint4 pa = *(const uint4*)(wi + k);
        uint4 pc = *(const uint4*)(wh + k);
        s = fmaf(bflo(pa.x), ha[k+0], s); s = fmaf(bfhi(pa.x), ha[k+1], s);
        s = fmaf(bflo(pa.y), ha[k+2], s); s = fmaf(bfhi(pa.y), ha[k+3], s);
        s = fmaf(bflo(pa.z), ha[k+4], s); s = fmaf(bfhi(pa.z), ha[k+5], s);
        s = fmaf(bflo(pa.w), ha[k+6], s); s = fmaf(bfhi(pa.w), ha[k+7], s);
        s = fmaf(bflo(pc.x), hb[k+0], s); s = fmaf(bfhi(pc.x), hb[k+1], s);
        s = fmaf(bflo(pc.y), hb[k+2], s); s = fmaf(bfhi(pc.y), hb[k+3], s);
        s = fmaf(bflo(pc.z), hb[k+4], s); s = fmaf(bfhi(pc.z), hb[k+5], s);
        s = fmaf(bflo(pc.w), hb[k+6], s); s = fmaf(bfhi(pc.w), hb[k+7], s);
      }
      s += __shfl_xor(s, 1); s += __shfl_xor(s, 2);
      if (qq == 0) {
        gv[ridx] = s + b2s[ridx];
        if (n + 1 < NSEQ) __builtin_prefetch(&P1[(size_t)(n + 1) * G4 + rglob], 0, 1);
      }
    }
    __syncthreads();
    if (tid < 16) {
      int j = wg * 16 + tid;
      float gi = gv[tid], gf = gv[16 + tid], gc = gv[32 + tid], go = gv[48 + tid];
      float c2n = sigm(gf) * c2 + sigm(gi) * tanhf(gc);
      float h2n = sigm(go) * tanhf(c2n);
      bool m = (n & 31) < (lengths[n >> 5] - 1);
      c2 = m ? c2n : c2;
      float hv = m ? h2n : h2s[j];
      h2g[qb + j] = hv;
      Ybf[(size_t)n * H_ + j] = f2bf(hv);   // ys[n] = carried h2 (bf16 for logits WMMA)
    }
    __threadfence();
    bar += NWG; grid_sync(cnt, bar);
  }
}

// ---------------- launch ----------------
extern "C" void kernel_launch(void* const* d_in, const int* in_sizes, int n_in,
                              void* d_out, int out_size, void* d_ws, size_t ws_size,
                              hipStream_t stream) {
  (void)in_sizes; (void)n_in; (void)out_size; (void)ws_size;
  const float* story    = (const float*)d_in[0];
  const int*   captions = (const int*)  d_in[1];
  const int*   lengths  = (const int*)  d_in[2];
  const float* W_story  = (const float*)d_in[3];
  const float* b_story  = (const float*)d_in[4];
  const float* embed    = (const float*)d_in[5];
  const float* W_ih1    = (const float*)d_in[6];
  const float* W_hh1    = (const float*)d_in[7];
  const float* b1       = (const float*)d_in[8];
  const float* W_ih2    = (const float*)d_in[9];
  const float* W_hh2    = (const float*)d_in[10];
  const float* b2       = (const float*)d_in[11];
  const float* W_out    = (const float*)d_in[12];
  const float* b_out    = (const float*)d_in[13];
  float* out = (float*)d_out;

  char* ws = (char*)d_ws;
  size_t off = 0;
  auto alloc = [&](size_t bytes) -> char* {
    char* p = ws + off;
    off = (off + bytes + 255) & ~(size_t)255;
    return p;
  };
  float* feats  = (float*)alloc((size_t)B_ * H_ * 4);
  u16* Xbf      = (u16*) alloc((size_t)NSEQ * D1_ * 2);
  u16* Wih1bf   = (u16*) alloc((size_t)G4 * D1_ * 2);
  float* P1     = (float*)alloc((size_t)NSEQ * G4 * 4);
  u16* Whh1bf   = (u16*) alloc((size_t)G4 * H_ * 2);
  u16* Wih2bf   = (u16*) alloc((size_t)G4 * H_ * 2);
  u16* Whh2bf   = (u16*) alloc((size_t)G4 * H_ * 2);
  u16* Woutbf   = (u16*) alloc((size_t)VPAD * H_ * 2);
  u16* Ybf      = (u16*) alloc((size_t)NSEQ * H_ * 2);
  float* h1g    = (float*)alloc(2 * H_ * 4);
  float* h2g    = (float*)alloc(2 * H_ * 4);
  u32* cnt      = (u32*) alloc(256);

  const size_t SMEM = (size_t)3 * 64 * 512 * 2 + (512 + 512 + 64 + 64) * 4; // 201216 B
  (void)hipFuncSetAttribute((const void*)lstm_scan_kernel,
                            hipFuncAttributeMaxDynamicSharedMemorySize, (int)SMEM);

  // Output start rows + sync/state init
  start_rows_kernel<<<(B_ * V_ + 255) / 256, 256, 0, stream>>>(out);
  init_sync_kernel<<<1, 256, 0, stream>>>(cnt, h1g, h2g);

  // feats, x build, bf16 conversions
  feats_kernel<<<(B_ * H_) / 256, 256, 0, stream>>>(story, W_story, b_story, feats);
  build_x_kernel<<<(NSEQ * D1_) / 256, 256, 0, stream>>>(feats, embed, captions, Xbf);
  cvt_bf_kernel<<<(G4 * D1_) / 256, 256, 0, stream>>>(W_ih1, Wih1bf, G4 * D1_);
  cvt_bf_kernel<<<(G4 * H_) / 256, 256, 0, stream>>>(W_hh1, Whh1bf, G4 * H_);
  cvt_bf_kernel<<<(G4 * H_) / 256, 256, 0, stream>>>(W_ih2, Wih2bf, G4 * H_);
  cvt_bf_kernel<<<(G4 * H_) / 256, 256, 0, stream>>>(W_hh2, Whh2bf, G4 * H_);
  wout_pad_kernel<<<(VPAD * H_) / 256, 256, 0, stream>>>(W_out, Woutbf);

  // Hoisted input projection (WMMA): P1 = X * W_ih1^T + b1
  proj1_wmma_kernel<<<dim3(G4 / 64, NSEQ / 32), 32, 0, stream>>>(Xbf, Wih1bf, b1, P1);

  // Sequential 2048-step scan, persistent 32-WG grid with LDS-resident weights (TDM preload)
  lstm_scan_kernel<<<NWG, STPB, SMEM, stream>>>(P1, Whh1bf, Wih2bf, Whh2bf, b2,
                                                lengths, h1g, h2g, Ybf, cnt);

  // Logits (WMMA) + shift/mask store into d_out
  logits_wmma_kernel<<<dim3(VPAD / 64, NSEQ / 32), 32, 0, stream>>>(Ybf, Woutbf, b_out,
                                                                    lengths, out);
}
